// AutoregressiveConvLSTM_22290880266388
// MI455X (gfx1250) — matrix operand order repaired and checked by reading, not compiled, not run on hardware
//
#include <hip/hip_runtime.h>
#include <math.h>

#define Bsz 32
#define Cch 16
#define Hdim 64
#define Wdim 64
#define HIDc 64
#define LOG2PI_F 1.8378770664093453f

typedef _Float16 half16v __attribute__((ext_vector_type(16)));
typedef _Float16 half8v  __attribute__((ext_vector_type(8)));
typedef float    f32x8   __attribute__((ext_vector_type(8)));

union AU { half16v v; half8v h[2]; };

__device__ __forceinline__ float sigf(float x) { return 1.0f / (1.0f + expf(-x)); }

// ---------------------------------------------------------------------------
// Init: zero c-state (f32), both h buffers (f16), and d_out (32 floats).
// ---------------------------------------------------------------------------
__global__ __launch_bounds__(256) void zero_init_kernel(
    float* __restrict__ c_st, _Float16* __restrict__ h0,
    _Float16* __restrict__ h1, float* __restrict__ out)
{
    const unsigned i = blockIdx.x * 256u + threadIdx.x;
    const unsigned N = (unsigned)Bsz * Hdim * Wdim * HIDc; // 8388608
    if (i < N) {
        c_st[i] = 0.0f;
        h0[i] = (_Float16)0.0f;
        h1[i] = (_Float16)0.0f;
    }
    if (i < 32u) out[i] = 0.0f;
}

// ---------------------------------------------------------------------------
// Pack w_ih / w_hh (HWIO f32, [tap][cin][cout]) -> f16 [tap][cout][cin]
// (B-matrix friendly: 16 contiguous cin per lane).
// ---------------------------------------------------------------------------
__global__ __launch_bounds__(256) void pack_w_kernel(
    const float* __restrict__ w_ih, const float* __restrict__ w_hh,
    _Float16* __restrict__ p_ih, _Float16* __restrict__ p_hh)
{
    const int i = blockIdx.x * 256 + threadIdx.x; // 0..147455
    if (i >= 9 * 256 * 64) return;
    const int tap = i / (256 * 64);
    const int r   = i - tap * (256 * 64);
    const int co  = r >> 6;
    const int ci  = r & 63;
    const int src = (tap * 64 + ci) * 256 + co;
    p_ih[i] = (_Float16)w_ih[src];
    p_hh[i] = (_Float16)w_hh[src];
}

// ---------------------------------------------------------------------------
// Step kernel 1: h_in = conv3x3(prev_x[1ch] -> 64ch) + b_in, written as f16.
// prev_x = x[:, t-1] (NCHW), zero when t == 0.  One thread per pixel.
// ---------------------------------------------------------------------------
__global__ __launch_bounds__(256) void input_conv_kernel(
    const float* __restrict__ x, const float* __restrict__ w_in,
    const float* __restrict__ b_in, _Float16* __restrict__ h_in, int t)
{
    const int pix = blockIdx.x * 256 + threadIdx.x; // B*H*W = 131072
    const int b  = pix >> 12;
    const int y  = (pix >> 6) & 63;
    const int xw = pix & 63;

    float px[9];
    #pragma unroll
    for (int kh = 0; kh < 3; ++kh) {
        #pragma unroll
        for (int kw = 0; kw < 3; ++kw) {
            const int sy = y + kh - 1, sx = xw + kw - 1;
            float v = 0.0f;
            if (t > 0 && sy >= 0 && sy < Hdim && sx >= 0 && sx < Wdim)
                v = x[(((b * Cch) + (t - 1)) * Hdim + sy) * Wdim + sx];
            px[kh * 3 + kw] = v;
        }
    }
    _Float16* op = h_in + (size_t)pix * HIDc;
    #pragma unroll
    for (int cc = 0; cc < 4; ++cc) {
        float acc[16];
        #pragma unroll
        for (int j = 0; j < 16; ++j) acc[j] = b_in[cc * 16 + j];
        #pragma unroll
        for (int tap = 0; tap < 9; ++tap) {
            #pragma unroll
            for (int j = 0; j < 16; ++j)
                acc[j] += px[tap] * w_in[tap * 64 + cc * 16 + j];
        }
        #pragma unroll
        for (int j = 0; j < 16; ++j) op[cc * 16 + j] = (_Float16)acc[j];
    }
}

// ---------------------------------------------------------------------------
// Step kernel 2 (WMMA workhorse): gates = conv3x3(h_in, w_ih) +
// conv3x3(h_prev, w_hh) + biases, then LSTM cell update.
//
// Blocking:
//  * Block: one hidden-channel group cg (16 channels, all 4 gates) x 256
//    pixels. Weights for this cg (2 convs x 9 taps x 4 gates x 16 couts x
//    64 cin, f16 = 147456 B) are staged once into LDS and shared by all
//    8 waves -> ~16x less global weight traffic, B feeds from ds_load_b128.
//  * Wave: 2 M-tiles (32 pixels in one row) x 4 gate tiles. Each B tile read
//    from LDS feeds 2 independent WMMAs (hazard hiding + LDS reuse).
//  Static WMMA count: 2 conv x 9 tap x 2 kc x 4 gate x 2 M = 288.
// ---------------------------------------------------------------------------
__global__ __launch_bounds__(256) void gates_lstm_kernel(
    const _Float16* __restrict__ h_in,  const _Float16* __restrict__ h_prev,
    const _Float16* __restrict__ wp_ih, const _Float16* __restrict__ wp_hh,
    const float* __restrict__ b_ih,     const float* __restrict__ b_hh,
    float* __restrict__ c_state,        _Float16* __restrict__ h_out)
{
    __shared__ _Float16 wlds[2 * 9 * 4 * 16 * 64]; // 147456 B

    const int lane = threadIdx.x & 31;
    const int ln   = lane & 15;   // pixel row within M-tile / N within tile
    const int hi   = lane >> 4;   // half-wave select
    const int wv   = threadIdx.x >> 5;

    const int cg   = blockIdx.x & 3;      // hidden-channel group (16 ch)
    const int pblk = blockIdx.x >> 2;     // 512 pixel-blocks of 256 pixels
    const int b    = pblk >> 4;           // 16 pixel-blocks per batch image
    const int rg   = pblk & 15;           // row group (4 rows)
    const int y    = rg * 4 + (wv >> 1);  // H row of this wave
    const int w0   = (wv & 1) << 5;       // column base: 0 or 32

    // ---- cooperative weight staging: global (packed f16) -> LDS ----------
    for (int c = threadIdx.x; c < 9216; c += 256) {   // 9216 chunks of 8 halfs
        const int half_off = c * 8;
        const int row = c >> 3;           // 0..1151 = ((cv*9+tap)*4+g)*16+rr
        const int ci8 = half_off & 63;
        const int cv  = row / 576;
        const int r1  = row - cv * 576;
        const int tap = r1 >> 6;
        const int r2  = r1 & 63;
        const int g   = r2 >> 4;
        const int rr  = r2 & 15;
        const _Float16* wp = cv ? wp_hh : wp_ih;
        const size_t soff =
            ((size_t)(tap * 256 + g * 64 + cg * 16 + rr)) * 64 + ci8;
        *(half8v*)(wlds + half_off) = *(const half8v*)(wp + soff);
    }
    __syncthreads();

    // ---- accumulators: [mtile][gate], bias-initialized --------------------
    f32x8 acc[2][4];
    {
        const int chb = cg * 16 + ln;
        #pragma unroll
        for (int g = 0; g < 4; ++g) {
            const float bb = b_ih[g * 64 + chb] + b_hh[g * 64 + chb];
            f32x8 v;
            #pragma unroll
            for (int r = 0; r < 8; ++r) v[r] = bb;
            acc[0][g] = v;
            acc[1][g] = v;
        }
    }

    const int base_b = b * Hdim;
    #pragma unroll
    for (int cv = 0; cv < 2; ++cv) {
        const _Float16* __restrict__ src = cv ? h_prev : h_in;
        #pragma unroll
        for (int kh = 0; kh < 3; ++kh) {
            const int sy = y + kh - 1;
            if (sy < 0 || sy >= Hdim) continue;      // uniform across wave
            #pragma unroll
            for (int kw = 0; kw < 3; ++kw) {
                const int tap = kh * 3 + kw;
                #pragma unroll
                for (int kc = 0; kc < 2; ++kc) {
                    AU a[2];
                    #pragma unroll
                    for (int mt = 0; mt < 2; ++mt) {
                        const int sx = w0 + mt * 16 + ln + kw - 1;
                        const bool inb = (sx >= 0) && (sx < Wdim);
                        if (inb) {
                            const _Float16* p =
                                src + ((size_t)((base_b + sy) * Wdim + sx)) * HIDc
                                    + kc * 32 + hi * 16;
                            a[mt].h[0] = *(const half8v*)p;        // K 16h+0..7
                            a[mt].h[1] = *(const half8v*)(p + 8);  // K 16h+8..15
                        } else {
                            a[mt].v = half16v{};
                        }
                    }
                    #pragma unroll
                    for (int g = 0; g < 4; ++g) {
                        const half16v bv = *(const half16v*)(
                            wlds + (size_t)(((cv * 9 + tap) * 4 + g) * 16 + ln) * 64
                                 + kc * 32 + hi * 16);
                        #pragma unroll
                        for (int mt = 0; mt < 2; ++mt) {
                            acc[mt][g] = __builtin_amdgcn_wmma_f32_16x16x32_f16(
                                false, a[mt].v, false, bv, (short)0,
                                acc[mt][g], false, false);
                        }
                    }
                }
            }
        }
    }

    // ---- LSTM update: lane owns channel cg*16+ln, pixels m = r + 8*hi -----
    #pragma unroll
    for (int mt = 0; mt < 2; ++mt) {
        const size_t pbase =
            ((size_t)(base_b + y) * Wdim + (w0 + mt * 16)) * HIDc + (cg * 16 + ln);
        float*    cp = c_state + pbase;
        _Float16* hp = h_out   + pbase;
        #pragma unroll
        for (int r = 0; r < 8; ++r) {
            const int m  = r + 8 * hi;
            const float gi = acc[mt][0][r];
            const float gg = acc[mt][1][r];
            const float gf = acc[mt][2][r];
            const float go = acc[mt][3][r];
            const float c_old = cp[(size_t)m * HIDc];
            const float c_new = sigf(gf + 1.0f) * c_old + sigf(gi) * tanhf(gg);
            const float h_new = sigf(go) * tanhf(c_new);
            cp[(size_t)m * HIDc] = c_new;
            hp[(size_t)m * HIDc] = (_Float16)h_new;
        }
    }
}

// ---------------------------------------------------------------------------
// Step kernel 3: (mu, log_sigma) = conv3x3(h_new, w_out) + b_out; Gaussian
// log-prob per pixel; block reduce; write per-block partial (deterministic).
// Grid: B*16 blocks x 256 threads (one thread per pixel).
// ---------------------------------------------------------------------------
__global__ __launch_bounds__(256) void out_conv_lp_kernel(
    const _Float16* __restrict__ hbuf, const float* __restrict__ w_out,
    const float* __restrict__ b_out,   const float* __restrict__ x,
    int t, float* __restrict__ partial)
{
    __shared__ float red[256];
    const int b   = blockIdx.x >> 4;
    const int pin = ((blockIdx.x & 15) << 8) + threadIdx.x; // 0..4095
    const int y   = pin >> 6;
    const int xw  = pin & 63;

    float mu = b_out[0];
    float ls = b_out[1];
    for (int kh = 0; kh < 3; ++kh) {
        const int sy = y + kh - 1;
        if (sy < 0 || sy >= Hdim) continue;
        for (int kw = 0; kw < 3; ++kw) {
            const int sx = xw + kw - 1;
            if (sx < 0 || sx >= Wdim) continue;
            const _Float16* hpx =
                hbuf + ((size_t)((b * Hdim + sy) * Wdim + sx)) * HIDc;
            const float* wp = w_out + (kh * 3 + kw) * 128;
            #pragma unroll 8
            for (int ci = 0; ci < 64; ++ci) {
                const float hv = (float)hpx[ci];
                mu += hv * wp[ci * 2 + 0];
                ls += hv * wp[ci * 2 + 1];
            }
        }
    }
    const float cur = x[(((b * Cch) + t) * Hdim + y) * Wdim + xw];
    const float z   = (cur - mu) * expf(-ls);
    const float lp  = -0.5f * z * z - ls - 0.5f * LOG2PI_F;

    red[threadIdx.x] = lp;
    __syncthreads();
    for (int s = 128; s > 0; s >>= 1) {
        if ((int)threadIdx.x < s) red[threadIdx.x] += red[threadIdx.x + s];
        __syncthreads();
    }
    if (threadIdx.x == 0) partial[blockIdx.x] = red[0];
}

// ---------------------------------------------------------------------------
// Deterministic accumulate of the 16 per-block partials into d_out[b].
// ---------------------------------------------------------------------------
__global__ void accum_kernel(const float* __restrict__ partial,
                             float* __restrict__ out)
{
    const int b = threadIdx.x;
    if (b < 32) {
        float s = 0.0f;
        #pragma unroll
        for (int j = 0; j < 16; ++j) s += partial[b * 16 + j];
        out[b] += s;
    }
}

// ---------------------------------------------------------------------------
// Launch
// ---------------------------------------------------------------------------
extern "C" void kernel_launch(void* const* d_in, const int* in_sizes, int n_in,
                              void* d_out, int out_size, void* d_ws, size_t ws_size,
                              hipStream_t stream)
{
    const float* x     = (const float*)d_in[0];
    const float* w_in  = (const float*)d_in[1];
    const float* b_in  = (const float*)d_in[2];
    const float* w_ih  = (const float*)d_in[3];
    const float* b_ih  = (const float*)d_in[4];
    const float* w_hh  = (const float*)d_in[5];
    const float* b_hh  = (const float*)d_in[6];
    const float* w_out = (const float*)d_in[7];
    const float* b_out = (const float*)d_in[8];
    float* out = (float*)d_out;

    char* ws = (char*)d_ws;
    // Workspace layout (bytes):
    //   c-state f32 : 33,554,432
    //   h buf 0 f16 : 16,777,216
    //   h buf 1 f16 : 16,777,216
    //   h_in    f16 : 16,777,216
    //   wp_ih   f16 :    294,912
    //   wp_hh   f16 :    294,912
    //   partials f32:      2,048
    float*    c_st  = (float*)(ws);
    _Float16* hbuf0 = (_Float16*)(ws + 33554432u);
    _Float16* hbuf1 = (_Float16*)(ws + 50331648u);
    _Float16* h_in  = (_Float16*)(ws + 67108864u);
    _Float16* wpih  = (_Float16*)(ws + 83886080u);
    _Float16* wphh  = (_Float16*)(ws + 84180992u);
    float*    part  = (float*)(ws + 84475904u);

    zero_init_kernel<<<32768, 256, 0, stream>>>(c_st, hbuf0, hbuf1, out);
    pack_w_kernel<<<576, 256, 0, stream>>>(w_ih, w_hh, wpih, wphh);

    for (int t = 0; t < 16; ++t) {
        _Float16* hprev = (t & 1) ? hbuf1 : hbuf0;
        _Float16* hnew  = (t & 1) ? hbuf0 : hbuf1;
        input_conv_kernel<<<512, 256, 0, stream>>>(x, w_in, b_in, h_in, t);
        gates_lstm_kernel<<<2048, 256, 0, stream>>>(h_in, hprev, wpih, wphh,
                                                    b_ih, b_hh, c_st, hnew);
        out_conv_lp_kernel<<<512, 256, 0, stream>>>(hnew, w_out, b_out, x, t, part);
        accum_kernel<<<1, 32, 0, stream>>>(part, out);
    }
}
